// TI_GCNNoEdgeAttrs_23244363006452
// MI455X (gfx1250) — matrix operand order repaired and checked by reading, not compiled, not run on hardware
//
#include <hip/hip_runtime.h>
#include <hip/hip_bf16.h>
#include <stdint.h>

typedef __attribute__((ext_vector_type(16))) __bf16 v16bf;
typedef __attribute__((ext_vector_type(8)))  float  v8f;

#define NEG_SLOPE 0.01f
#define NUM_GRAPHS 64

__device__ __forceinline__ float lrelu(float v) { return v > 0.0f ? v : NEG_SLOPE * v; }

// float -> bf16 with round-to-nearest-even (inputs are finite)
__device__ __forceinline__ unsigned short f2bf(float f) {
    unsigned u = __float_as_uint(f);
    unsigned r = u + 0x7FFFu + ((u >> 16) & 1u);
    return (unsigned short)(r >> 16);
}

// ---------------------------------------------------------------- conversions
__global__ void cvt_bf16_kernel(const float* __restrict__ in,
                                unsigned short* __restrict__ out, long long n) {
    long long i = (long long)blockIdx.x * blockDim.x + threadIdx.x;
    if (i < n) out[i] = f2bf(in[i]);
}

// Pack weight matrix W[K,Fout] (row-major f32) into WMMA B-fragment layout:
// per (k-tile kt, n-tile nt): 32 lanes x 16 bf16, element i of lane L is
//   W[(kt*32 + (L<16?0:8) + (i<8 ? i : i+8)) * Fout + nt*16 + (L&15)]
// stored contiguously so a wave fetches its fragment as 2 x b128 per lane.
__global__ void pack_w_kernel(const float* __restrict__ W,
                              unsigned short* __restrict__ P,
                              int Fout, int ntiles, int total) {
    int gid = blockIdx.x * blockDim.x + threadIdx.x;
    if (gid >= total) return;
    int i  = gid & 15;
    int L  = (gid >> 4) & 31;
    int t  = gid >> 9;              // tile index = kt*ntiles + nt
    int nt = t % ntiles;
    int kt = t / ntiles;
    int k  = kt * 32 + ((L < 16) ? 0 : 8) + ((i < 8) ? i : i + 8);
    int n  = nt * 16 + (L & 15);
    P[gid] = f2bf(W[(size_t)k * Fout + n]);
}

// ---------------------------------------------------------------- degree / dinv
__global__ void deg_kernel(const long long* __restrict__ dst,
                           float* __restrict__ deg, int E) {
    int e = blockIdx.x * blockDim.x + threadIdx.x;
    if (e < E) atomicAdd(&deg[(int)dst[e]], 1.0f);
}

__global__ void dinv_kernel(float* __restrict__ deg, int N) {
    int i = blockIdx.x * blockDim.x + threadIdx.x;
    if (i < N) deg[i] = rsqrtf(deg[i] + 1.0f);   // deg includes self-loop (+1)
}

// ---------------------------------------------------------------- GEMM1: [N,128] x [128,64] -> [N,64]
// block = 128 threads (4 waves); wave w computes 16x16 tile at (blockIdx*16, w*16)
__global__ void __launch_bounds__(128)
gemm1_kernel(const unsigned short* __restrict__ xb,   // bf16 [N,128]
             const unsigned short* __restrict__ w1p,  // packed B fragments (4 ktiles x 4 ntiles)
             float* __restrict__ h1, int N) {
    const int lane = threadIdx.x & 31;
    const int wave = threadIdx.x >> 5;
    const int m0   = blockIdx.x * 16;
    const int col  = lane & 15;
    const int rowA = m0 + col;
    const int rA   = rowA < N ? rowA : N - 1;
    const int kbA  = (lane < 16) ? 0 : 8;     // A: lanes 16-31 hold the K+8 group

    v8f acc = {};
#pragma unroll
    for (int k0 = 0; k0 < 128; k0 += 32) {
        union { uint4 q[2]; v16bf v; } a;
        a.q[0] = *(const uint4*)(xb + (size_t)rA * 128 + k0 + kbA);
        a.q[1] = *(const uint4*)(xb + (size_t)rA * 128 + k0 + 16 + kbA);
        union { uint4 q[2]; v16bf v; } b;
        const unsigned short* bp = w1p + (((size_t)(k0 >> 5) * 4 + wave) * 32 + lane) * 16;
        b.q[0] = *(const uint4*)(bp);
        b.q[1] = *(const uint4*)(bp + 8);
        acc = __builtin_amdgcn_wmma_f32_16x16x32_bf16(
                  false, a.v, false, b.v, (short)0, acc, false, false);
    }
    const int n0    = wave * 16;
    const int rbase = m0 + ((lane < 16) ? 0 : 8);
    if (m0 + 16 <= N) {                       // full tile: straight-line stores
        float* p = h1 + (size_t)rbase * 64 + n0 + col;
#pragma unroll
        for (int i = 0; i < 8; ++i) p[(size_t)i * 64] = acc[i];
    } else {                                  // tail tile (not hit when N % 16 == 0)
#pragma unroll
        for (int i = 0; i < 8; ++i) {
            int row = rbase + i;
            if (row < N) h1[(size_t)row * 64 + n0 + col] = acc[i];
        }
    }
}

// ---------------------------------------------------------------- edge scatter, layer 1 (64 features)
// one wave per edge: 2 floats per lane; index loads are wave-uniform (single txn)
__global__ void msg1_kernel(const long long* __restrict__ src,
                            const long long* __restrict__ dst,
                            const float* __restrict__ dinv,
                            const float* __restrict__ h1,
                            float* __restrict__ agg, int E) {
    long long gid = (long long)blockIdx.x * blockDim.x + threadIdx.x;
    int e = (int)(gid >> 5);
    int l = (int)(gid & 31);
    if (e >= E) return;
    int s = (int)src[e], d = (int)dst[e];
    float norm = dinv[s] * dinv[d];
    float2 v = *(const float2*)(h1 + (size_t)s * 64 + l * 2);
    float* p = agg + (size_t)d * 64 + l * 2;
    atomicAdd(p,     v.x * norm);
    atomicAdd(p + 1, v.y * norm);
}

// self-loop + bias + leaky-relu, emit bf16 (feeds GEMM2 only)
__global__ void finish1_kernel(const float* __restrict__ agg,
                               const float* __restrict__ h1,
                               const float* __restrict__ dinv,
                               const float* __restrict__ b1,
                               unsigned short* __restrict__ a1b, int N) {
    long long gid = (long long)blockIdx.x * blockDim.x + threadIdx.x;
    if (gid >= (long long)N * 64) return;
    int i = (int)(gid >> 6);
    int f = (int)(gid & 63);
    float di = dinv[i];
    float v = agg[gid] + h1[gid] * di * di + b1[f];
    a1b[gid] = f2bf(lrelu(v));
}

// ---------------------------------------------------------------- GEMM2: [N,64] x [64,16] -> [N,16]
// block = 64 threads (2 waves); each wave owns one 16-row tile (single N-tile)
__global__ void __launch_bounds__(64)
gemm2_kernel(const unsigned short* __restrict__ ab,   // bf16 [N,64]
             const unsigned short* __restrict__ w2p,  // packed B fragments (2 ktiles x 1 ntile)
             float* __restrict__ h2, int N) {
    const int lane = threadIdx.x & 31;
    const int wave = threadIdx.x >> 5;
    const int m0   = (blockIdx.x * 2 + wave) * 16;
    const int col  = lane & 15;
    const int rowA = m0 + col;
    const int rA   = rowA < N ? rowA : N - 1;
    const int kbA  = (lane < 16) ? 0 : 8;

    v8f acc = {};
#pragma unroll
    for (int k0 = 0; k0 < 64; k0 += 32) {
        union { uint4 q[2]; v16bf v; } a;
        a.q[0] = *(const uint4*)(ab + (size_t)rA * 64 + k0 + kbA);
        a.q[1] = *(const uint4*)(ab + (size_t)rA * 64 + k0 + 16 + kbA);
        union { uint4 q[2]; v16bf v; } b;
        const unsigned short* bp = w2p + ((size_t)(k0 >> 5) * 32 + lane) * 16;
        b.q[0] = *(const uint4*)(bp);
        b.q[1] = *(const uint4*)(bp + 8);
        acc = __builtin_amdgcn_wmma_f32_16x16x32_bf16(
                  false, a.v, false, b.v, (short)0, acc, false, false);
    }
    const int rbase = m0 + ((lane < 16) ? 0 : 8);
    if (m0 + 16 <= N) {
        float* p = h2 + (size_t)rbase * 16 + col;
#pragma unroll
        for (int i = 0; i < 8; ++i) p[(size_t)i * 16] = acc[i];
    } else {
#pragma unroll
        for (int i = 0; i < 8; ++i) {
            int row = rbase + i;
            if (row < N) h2[(size_t)row * 16 + col] = acc[i];
        }
    }
}

// ---------------------------------------------------------------- edge scatter, layer 2 (16 features)
__global__ void msg2_kernel(const long long* __restrict__ src,
                            const long long* __restrict__ dst,
                            const float* __restrict__ dinv,
                            const float* __restrict__ h2,
                            float* __restrict__ agg, int E) {
    long long gid = (long long)blockIdx.x * blockDim.x + threadIdx.x;
    int e = (int)(gid >> 4);
    int f = (int)(gid & 15);
    if (e >= E) return;
    int s = (int)src[e], d = (int)dst[e];
    float norm = dinv[s] * dinv[d];
    atomicAdd(&agg[(size_t)d * 16 + f], h2[(size_t)s * 16 + f] * norm);
}

// finish layer 2 + pooled sums/counts
__global__ void finish2_pool_kernel(const float* __restrict__ agg,
                                    const float* __restrict__ h2,
                                    const float* __restrict__ dinv,
                                    const float* __restrict__ b2,
                                    const long long* __restrict__ batch,
                                    float* __restrict__ psum,
                                    float* __restrict__ pcnt, int N) {
    long long gid = (long long)blockIdx.x * blockDim.x + threadIdx.x;
    if (gid >= (long long)N * 16) return;
    int i = (int)(gid >> 4);
    int f = (int)(gid & 15);
    float di = dinv[i];
    float v = lrelu(agg[gid] + h2[gid] * di * di + b2[f]);
    int g = (int)batch[i];
    atomicAdd(&psum[g * 16 + f], v);
    if (f == 0) atomicAdd(&pcnt[g], 1.0f);
}

// ---------------------------------------------------------------- MLP head: mean -> 16->32 lrelu -> 32->10
__global__ void head_kernel(const float* __restrict__ psum,
                            const float* __restrict__ pcnt,
                            const float* __restrict__ fc1_W, const float* __restrict__ fc1_b,
                            const float* __restrict__ fc3_W, const float* __restrict__ fc3_b,
                            float* __restrict__ out) {
    int t = threadIdx.x;
    if (t >= NUM_GRAPHS) return;
    float cnt = fmaxf(pcnt[t], 1.0f);
    float g[16];
#pragma unroll
    for (int f = 0; f < 16; ++f) g[f] = psum[t * 16 + f] / cnt;
    float z[32];
#pragma unroll
    for (int j = 0; j < 32; ++j) {
        float s = fc1_b[j];
        for (int i = 0; i < 16; ++i) s += g[i] * fc1_W[i * 32 + j];
        z[j] = lrelu(s);
    }
#pragma unroll
    for (int c = 0; c < 10; ++c) {
        float s = fc3_b[c];
        for (int i = 0; i < 32; ++i) s += z[i] * fc3_W[i * 10 + c];
        out[t * 10 + c] = s;
    }
}

// ================================================================ launch
extern "C" void kernel_launch(void* const* d_in, const int* in_sizes, int n_in,
                              void* d_out, int out_size, void* d_ws, size_t ws_size,
                              hipStream_t stream) {
    const float*     x     = (const float*)d_in[0];
    const long long* edges = (const long long*)d_in[1];
    const long long* batch = (const long long*)d_in[2];
    // d_in[3] = mask (unused by reference)
    const float* W1    = (const float*)d_in[4];
    const float* b1    = (const float*)d_in[5];
    const float* W2    = (const float*)d_in[6];
    const float* b2    = (const float*)d_in[7];
    const float* fc1_W = (const float*)d_in[8];
    const float* fc1_b = (const float*)d_in[9];
    const float* fc3_W = (const float*)d_in[10];
    const float* fc3_b = (const float*)d_in[11];
    float* out = (float*)d_out;

    const int N = in_sizes[2];          // nodes (batch vector length)
    const int E = in_sizes[1] / 2;      // edges
    const long long* srcI = edges;
    const long long* dstI = edges + E;

    // ---- workspace carve (region A = N*256 bytes is reused: xb -> a1b|h2|agg2)
    char* w = (char*)d_ws;
    const size_t NB = (size_t)N;
    unsigned short* xb   = (unsigned short*)w;                   // bf16 [N,128]  (dies after gemm1)
    unsigned short* a1b  = (unsigned short*)w;                   // bf16 [N,64]   (reuse)
    float*          h2   = (float*)(w + NB * 128);               // [N,16]
    float*          agg2 = (float*)(w + NB * 128 + NB * 64);     // [N,16]
    float*          h1   = (float*)(w + NB * 256);               // [N,64]
    float*          agg1 = (float*)(w + NB * 512);               // [N,64]
    float*          dinv = (float*)(w + NB * 768);               // [N]
    unsigned short* w1p  = (unsigned short*)(w + NB * 768 + NB * 4);  // packed W1 frags (8192)
    unsigned short* w2p  = w1p + 4 * 4 * 32 * 16;                      // packed W2 frags (1024)
    float*          psum = (float*)(w2p + 2 * 1 * 32 * 16);      // [64,16]
    float*          pcnt = psum + NUM_GRAPHS * 16;               // [64]

    // ---- zero accumulators (graph-capture-safe async memsets)
    hipMemsetAsync(dinv, 0, NB * 4, stream);
    hipMemsetAsync(agg1, 0, NB * 64 * 4, stream);
    hipMemsetAsync(agg2, 0, NB * 16 * 4, stream);
    hipMemsetAsync(psum, 0, (NUM_GRAPHS * 16 + NUM_GRAPHS) * 4, stream);

    // ---- bf16 conversion of activations + weight fragment packing
    {
        long long n = (long long)N * 128;
        cvt_bf16_kernel<<<(unsigned)((n + 255) / 256), 256, 0, stream>>>(x, xb, n);
        pack_w_kernel<<<(4 * 4 * 512 + 255) / 256, 256, 0, stream>>>(W1, w1p, 64, 4, 4 * 4 * 512);
        pack_w_kernel<<<(2 * 1 * 512 + 255) / 256, 256, 0, stream>>>(W2, w2p, 16, 1, 2 * 1 * 512);
    }

    // ---- degree + rsqrt
    deg_kernel<<<(E + 255) / 256, 256, 0, stream>>>(dstI, dinv, E);
    dinv_kernel<<<(N + 255) / 256, 256, 0, stream>>>(dinv, N);

    // ---- layer 1
    gemm1_kernel<<<(N + 15) / 16, 128, 0, stream>>>(xb, w1p, h1, N);
    {
        long long t = (long long)E * 32;
        msg1_kernel<<<(unsigned)((t + 255) / 256), 256, 0, stream>>>(srcI, dstI, dinv, h1, agg1, E);
    }
    {
        long long t = (long long)N * 64;
        finish1_kernel<<<(unsigned)((t + 255) / 256), 256, 0, stream>>>(agg1, h1, dinv, b1, a1b, N);
    }

    // ---- layer 2
    {
        int mtiles = (N + 15) / 16;
        gemm2_kernel<<<(mtiles + 1) / 2, 64, 0, stream>>>(a1b, w2p, h2, N);
    }
    {
        long long t = (long long)E * 16;
        msg2_kernel<<<(unsigned)((t + 255) / 256), 256, 0, stream>>>(srcI, dstI, dinv, h2, agg2, E);
    }
    {
        long long t = (long long)N * 16;
        finish2_pool_kernel<<<(unsigned)((t + 255) / 256), 256, 0, stream>>>(
            agg2, h2, dinv, b2, batch, psum, pcnt, N);
    }

    // ---- head
    head_kernel<<<1, 64, 0, stream>>>(psum, pcnt, fc1_W, fc1_b, fc3_W, fc3_b, out);
}
